// Time1dConvGLU_49323404427950
// MI455X (gfx1250) — matrix operand order repaired and checked by reading, not compiled, not run on hardware
//
#include <hip/hip_runtime.h>

typedef __attribute__((ext_vector_type(16))) _Float16 v16h;
typedef __attribute__((ext_vector_type(2)))  _Float16 v2h;
typedef __attribute__((ext_vector_type(8)))  float    v8f;

// x[N][64][64], w[128][64][3], bias[128] -> out[N][64][62]
// Per sample GEMM: A[128][192] x B[192][64pad], K = 192 = 6*32, f16 in / f32 acc.
#define SPW 8           // samples per (persistent) workgroup
#define XSTRIDE 68      // Xs row stride in floats (64 data + 4 zero pad, 16B-aligned rows)

__global__ __launch_bounds__(256) void conv1d_glu_wmma(
    const float* __restrict__ x,
    const float* __restrict__ w,
    const float* __restrict__ bias,
    float* __restrict__ out, int N)
{
    __shared__ float    Xs[64 * XSTRIDE];       // ~17 KB: one sample, zero-padded rows
    __shared__ _Float16 Wl[8 * 6 * 32 * 16];    // 48 KB: A fragments (WMMA lane layout)
    __shared__ _Float16 Xb[6 * 4 * 32 * 16];    // 24 KB: B fragments (im2col, lane layout)

    const int tid  = threadIdx.x;
    const int wv   = tid >> 5;          // wave id (wave32)
    const int lane = tid & 31;
    const int m    = wv & 3;            // M-pair (m, m+4) -> GLU a/b rows pair up in-lane
    const int np   = wv >> 2;           // N-pair (2np, 2np+1)
    const int g    = lane >> 4;
    const int nl   = lane & 15;

    // thread-invariant fragment coordinates (p = tid + 256*i => these don't depend on i)
    const int hp = tid & 7;             // half-pair index 0..7
    const int Lb = (tid >> 3) & 31;     // fragment lane 0..31
    const int gb = Lb >> 4;

    // ---- once per WG: zero-pad Xs columns 64..67 ----
    Xs[(tid >> 2) * XSTRIDE + 64 + (tid & 3)] = 0.0f;

    // ---- once per WG: build A fragments Wl[((m*6+kk)*32+L)*16+h] = f16(W[co][r]) ----
    // A 16x32 f16 layout: K = (h&7) + (h>=8?16:0) + 8*(L>>4), M = 16m + (L&15)
    {
        const int h0     = hp << 1;
        const int K0base = (h0 & 7) + ((hp >= 4) ? 16 : 0) + 8 * gb;
        const int coL    = Lb & 15;
        const float* wbase = w + coL * 192 + K0base;     // + 3072*mm + 32*kk, imm offsets
        _Float16* dst = &Wl[tid << 1];
#pragma unroll
        for (int i = 0; i < 48; ++i) {
            int kk = i % 6, mm = i / 6;                  // constants after unroll
            const float* wp = wbase + 3072 * mm + 32 * kk;
            v2h pair;
            pair[0] = (_Float16)wp[0];
            pair[1] = (_Float16)wp[1];
            *(v2h*)(dst + 512 * i) = pair;
        }
    }

    // ---- once per WG: per-lane bias (C layout: M = v + 8g within tile m / m+4) ----
    float bA[8], bB[8];
#pragma unroll
    for (int v = 0; v < 8; ++v) {
        int co = 16 * m + v + 8 * g;
        bA[v] = bias[co];
        bB[v] = bias[co + 64];
    }

    const int n0   = blockIdx.x * SPW;
    int       nEnd = n0 + SPW; if (nEnd > N) nEnd = N;

    // register prefetch of first sample (16 KB / 256 threads = 4 float4 each)
    float4 pf[4];
    if (n0 < N) {
        const float4* src = (const float4*)(x + (size_t)n0 * 4096);
#pragma unroll
        for (int i = 0; i < 4; ++i) pf[i] = src[tid + 256 * i];
    }

    for (int n = n0; n < nEnd; ++n) {
        // ---- stage prefetched sample into Xs (rows padded to XSTRIDE) ----
#pragma unroll
        for (int i = 0; i < 4; ++i) {
            int idx = tid + 256 * i;                     // float4 index 0..1023
            int row = idx >> 4;
            int col = (idx & 15) << 2;
            *(float4*)&Xs[row * XSTRIDE + col] = pf[i];
        }
        // barrier 1: Xs (+pad/Wl on iter 0) visible; also guarantees every thread
        // finished the previous GEMM, so Xb below is safe to overwrite.
        __syncthreads();

        // ---- build B fragments Xb[((kk*4+nt)*32+L)*16+h] = f16(x[ci][t+k]) ----
        // B 32x16 f16 layout: K = (h&7) + (h>=8?16:0) + 8*(L>>4), N = 16nt + (L&15); r = 32kk+K = ci*3+k
        {
            const int rbase = ((hp & 3) << 1) + ((hp >= 4) ? 16 : 0) + 8 * gb;       // thread-invariant
            const int tcb   = Lb & 15;
            _Float16* dst = &Xb[tid << 1];
#pragma unroll
            for (int i = 0; i < 24; ++i) {
                int nt = i & 3, kk = i >> 2;             // constants after unroll
                int tcol = 16 * nt + tcb;
                int r0   = 32 * kk + rbase;
                v2h pair;
#pragma unroll
                for (int j = 0; j < 2; ++j) {
                    int r  = r0 + j;
                    int ci = r / 3;
                    int k  = r - 3 * ci;
                    pair[j] = (_Float16)Xs[ci * XSTRIDE + tcol + k];  // pad cols are 0
                }
                *(v2h*)(dst + 512 * i) = pair;
            }
        }

        // prefetch next sample while im2col/LDS traffic drains
        if (n + 1 < nEnd) {
            const float4* src = (const float4*)(x + (size_t)(n + 1) * 4096);
#pragma unroll
            for (int i = 0; i < 4; ++i) pf[i] = src[tid + 256 * i];
        }
        __syncthreads();   // barrier 2: Xb visible

        // ---- GEMM: 6 K-steps, tiles (m,2np),(m,2np+1),(m+4,2np),(m+4,2np+1) ----
        v8f acc0 = {}, acc1 = {}, acc2 = {}, acc3 = {};
        const v16h* Wv = (const v16h*)Wl;
        const v16h* Bv = (const v16h*)Xb;
#pragma unroll
        for (int kk = 0; kk < 6; ++kk) {
            v16h a0 = Wv[(m * 6 + kk) * 32 + lane];
            v16h a1 = Wv[((m + 4) * 6 + kk) * 32 + lane];
            v16h b0 = Bv[(kk * 4 + 2 * np + 0) * 32 + lane];
            v16h b1 = Bv[(kk * 4 + 2 * np + 1) * 32 + lane];
            acc0 = __builtin_amdgcn_wmma_f32_16x16x32_f16(false, a0, false, b0, (short)0, acc0, false, false);
            acc1 = __builtin_amdgcn_wmma_f32_16x16x32_f16(false, a0, false, b1, (short)0, acc1, false, false);
            acc2 = __builtin_amdgcn_wmma_f32_16x16x32_f16(false, a1, false, b0, (short)0, acc2, false, false);
            acc3 = __builtin_amdgcn_wmma_f32_16x16x32_f16(false, a1, false, b1, (short)0, acc3, false, false);
        }

        // ---- bias + GLU + store: out[n][co][t], co = 16m+v+8g, t = 16*(2np+ni)+nl ----
        float* outn = out + (size_t)n * (64 * 62) + (size_t)(16 * m + 8 * g) * 62;
#pragma unroll
        for (int ni = 0; ni < 2; ++ni) {
            int t = 16 * (2 * np + ni) + nl;
            if (t < 62) {
                v8f aa = ni ? acc1 : acc0;
                v8f bb = ni ? acc3 : acc2;
#pragma unroll
                for (int v = 0; v < 8; ++v) {
                    float a = aa[v] + bA[v];
                    float b = bb[v] + bB[v];
                    float s = __builtin_amdgcn_rcpf(1.0f + __expf(-b));
                    outn[v * 62 + t] = a * s;
                }
            }
        }
    }
}

extern "C" void kernel_launch(void* const* d_in, const int* in_sizes, int n_in,
                              void* d_out, int out_size, void* d_ws, size_t ws_size,
                              hipStream_t stream) {
    const float* x    = (const float*)d_in[0];
    const float* w    = (const float*)d_in[1];
    const float* bias = (const float*)d_in[2];
    float* out = (float*)d_out;
    int N  = in_sizes[0] / (64 * 64);            // 10000
    int NB = (N + SPW - 1) / SPW;                // 1250 persistent WGs
    hipLaunchKernelGGL(conv1d_glu_wmma, dim3(NB), dim3(256), 0, stream, x, w, bias, out, N);
}
